// Model_50139448214015
// MI455X (gfx1250) — matrix operand order repaired and checked by reading, not compiled, not run on hardware
//
#include <hip/hip_runtime.h>

#define HC 256   // H*C = 4*64

typedef __attribute__((ext_vector_type(16))) __bf16 v16bf;
typedef __attribute__((ext_vector_type(8)))  float  v8f;

// ---------------------------------------------------------------- helpers
__device__ __forceinline__ void atomicMaxF(float* addr, float v) {
  if (v >= 0.f) atomicMax((int*)addr, __float_as_int(v));
  else          atomicMin((unsigned int*)addr, __float_as_uint(v));
}

// ---------------------------------------------------------------- lin1: tanh(x[N,64] @ W[64,8] + b)
__global__ void lin1_kernel(const float* __restrict__ x, const float* __restrict__ W,
                            const float* __restrict__ b, float* __restrict__ out, int N) {
  __shared__ float sW[64 * 8];
  __shared__ float sb[8];
  for (int i = threadIdx.x; i < 64 * 8; i += blockDim.x) sW[i] = W[i];
  if (threadIdx.x < 8) sb[threadIdx.x] = b[threadIdx.x];
  __syncthreads();
  int n = blockIdx.x * blockDim.x + threadIdx.x;
  if (n >= N) return;
  const float* xr = x + (size_t)n * 64;
  float acc[8];
#pragma unroll
  for (int o = 0; o < 8; ++o) acc[o] = sb[o];
  for (int k = 0; k < 64; ++k) {
    float xv = xr[k];
#pragma unroll
    for (int o = 0; o < 8; ++o) acc[o] += xv * sW[k * 8 + o];
  }
  float* orow = out + (size_t)n * 8;
#pragma unroll
  for (int o = 0; o < 8; ++o) orow[o] = tanhf(acc[o]);
}

// ---------------------------------------------------------------- layer-0 projection: x8[N,8] @ W[8,256] + b
__global__ void proj8_kernel(const float* __restrict__ x8, const float* __restrict__ W,
                             const float* __restrict__ b, float* __restrict__ out, int N) {
  __shared__ float sW[8 * HC];
  for (int i = threadIdx.x; i < 8 * HC; i += blockDim.x) sW[i] = W[i];
  __syncthreads();
  size_t idx = (size_t)blockIdx.x * blockDim.x + threadIdx.x;
  size_t n = idx >> 8;
  int c = (int)(idx & 255);
  if (n >= (size_t)N) return;
  const float* xr = x8 + n * 8;
  float acc = b[c];
#pragma unroll
  for (int k = 0; k < 8; ++k) acc += xr[k] * sW[k * HC + c];
  out[idx] = acc;
}

// ---------------------------------------------------------------- layer-1 projection: WMMA bf16, K=256, Ncols=256
// one wave per 16x64 output strip: A fragment loaded/converted once per K-step,
// reused across 4 column tiles (4 v_wmma per K-step, 32 total).
__global__ void gemm_bf16_wmma(const float* __restrict__ X,   // [M,256] f32
                               const float* __restrict__ W,   // [256,256] f32 row-major
                               const float* __restrict__ bias,// [256]
                               float* __restrict__ out, int M) {
  int wave = blockIdx.x * (blockDim.x >> 5) + (threadIdx.x >> 5);
  int lane = threadIdx.x & 31;
  int tm = wave >> 2;       // row tile (16 rows)
  int tg = wave & 3;        // 64-wide column group
  if (tm * 16 >= M) return; // wave-uniform; EXEC stays all-ones for WMMA
  const int m0 = tm << 4, n0 = tg << 6;
  const int row  = m0 + (lane & 15);        // A: lane -> M row
  const int koff = (lane & 16) >> 1;        // lanes 16-31 hold K+8 halves
  const float* xrow = X + (size_t)row * HC;
  v8f acc[4] = {};
#pragma unroll
  for (int k0 = 0; k0 < HC; k0 += 32) {
    // A fragment: elements 0..7 -> K = k0+koff+0..7 ; 8..15 -> K = k0+16+koff+0..7
    v16bf af;
    const float* xp = xrow + k0 + koff;
#pragma unroll
    for (int j = 0; j < 8; ++j) {
      af[j]     = (__bf16)xp[j];
      af[j + 8] = (__bf16)xp[j + 16];
    }
    // B fragments: lane&15 = N column of W, element = K row (same K order as A)
    const float* wp = W + (size_t)(k0 + koff) * HC + n0 + (lane & 15);
#pragma unroll
    for (int t = 0; t < 4; ++t) {
      v16bf bf;
#pragma unroll
      for (int j = 0; j < 8; ++j) {
        bf[j]     = (__bf16)wp[(size_t)j * HC + t * 16];
        bf[j + 8] = (__bf16)wp[(size_t)(j + 16) * HC + t * 16];
      }
      acc[t] = __builtin_amdgcn_wmma_f32_16x16x32_bf16(false, af, false, bf,
                                                       (short)0, acc[t], false, false);
    }
  }
  // D layout: VGPR r -> M = r (+8 for lanes 16-31), N = lane&15
  const int mrow = m0 + ((lane & 16) >> 1);
#pragma unroll
  for (int t = 0; t < 4; ++t) {
    const int col = n0 + t * 16 + (lane & 15);
    const float bb = bias[col];
#pragma unroll
    for (int r = 0; r < 8; ++r)
      out[(size_t)(mrow + r) * HC + col] = acc[t][r] + bb;
  }
}

// ---------------------------------------------------------------- segment-softmax init
__global__ void init_seg_kernel(float* __restrict__ mx, float* __restrict__ den, int n) {
  int i = blockIdx.x * blockDim.x + threadIdx.x;
  if (i < n) { mx[i] = -__builtin_inff(); den[i] = 0.f; }
}

__global__ void init_out_kernel(float* __restrict__ out, const float* __restrict__ b, size_t total) {
  size_t i = (size_t)blockIdx.x * blockDim.x + threadIdx.x;
  if (i < total) out[i] = b[i & 255];
}

// ---------------------------------------------------------------- per-edge logits + segment max (wave per edge)
__global__ void edge_logits_kernel(const float* __restrict__ xs, const float* __restrict__ xd,
                                   const float* __restrict__ ea, const float* __restrict__ We,
                                   const float* __restrict__ att,
                                   const int* __restrict__ src, const int* __restrict__ dst,
                                   float* __restrict__ logits, float* __restrict__ mx, int E) {
  int wid = blockIdx.x * (blockDim.x >> 5) + (threadIdx.x >> 5);
  int lane = threadIdx.x & 31;
  if (wid >= E) return;
  int s = src[wid], d = dst[wid];
  float a0 = ea[(size_t)wid * 3 + 0], a1 = ea[(size_t)wid * 3 + 1], a2 = ea[(size_t)wid * 3 + 2];
  const float* ps = xs + (size_t)s * HC + lane * 8;
  const float* pd = xd + (size_t)d * HC + lane * 8;
  const float* pw = We + lane * 8;       // We[3][256]
  const float* pa = att + lane * 8;      // att flat [4*64], channel i = h*64+c
  float lsum = 0.f;
#pragma unroll
  for (int j = 0; j < 8; ++j) {
    float e = a0 * pw[j] + a1 * pw[256 + j] + a2 * pw[512 + j];
    float g = ps[j] + pd[j] + e;
    g = g > 0.f ? g : 0.2f * g;          // leaky_relu 0.2
    lsum += g * pa[j];
  }
  // reduce within each 8-lane head group (64 channels / head)
  lsum += __shfl_xor(lsum, 1, 32);
  lsum += __shfl_xor(lsum, 2, 32);
  lsum += __shfl_xor(lsum, 4, 32);
  if ((lane & 7) == 0) {
    int h = lane >> 3;
    logits[(size_t)wid * 4 + h] = lsum;
    atomicMaxF(&mx[(size_t)d * 4 + h], lsum);
  }
}

// ---------------------------------------------------------------- exp + segment sum (thread per edge*head)
__global__ void edge_exp_kernel(float* __restrict__ logits, const float* __restrict__ mx,
                                float* __restrict__ den, const int* __restrict__ dst, int n) {
  int i = blockIdx.x * blockDim.x + threadIdx.x;
  if (i >= n) return;
  int e = i >> 2, h = i & 3;
  int d = dst[e];
  float al = __expf(logits[i] - mx[(size_t)d * 4 + h]);
  logits[i] = al;                               // overwrite with numerator
  unsafeAtomicAdd(&den[(size_t)d * 4 + h], al);
}

// ---------------------------------------------------------------- weighted scatter (wave per edge)
__global__ void edge_scatter_kernel(const float* __restrict__ al, const float* __restrict__ den,
                                    const float* __restrict__ xs,
                                    const int* __restrict__ src, const int* __restrict__ dst,
                                    float* __restrict__ out, int E) {
  int wid = blockIdx.x * (blockDim.x >> 5) + (threadIdx.x >> 5);
  int lane = threadIdx.x & 31;
  if (wid >= E) return;
  int s = src[wid], d = dst[wid];
  int h = lane >> 3;
  float alpha = al[(size_t)wid * 4 + h] / den[(size_t)d * 4 + h];
  const float* ps = xs + (size_t)s * HC + lane * 8;
  float* po = out + (size_t)d * HC + lane * 8;
#pragma unroll
  for (int j = 0; j < 8; ++j) unsafeAtomicAdd(&po[j], alpha * ps[j]);
}

// ---------------------------------------------------------------- elementwise tanh
__global__ void tanh_kernel(float* __restrict__ x, size_t n) {
  size_t i = (size_t)blockIdx.x * blockDim.x + threadIdx.x;
  if (i < n) x[i] = tanhf(x[i]);
}

// ---------------------------------------------------------------- actor head: [xm[src], ea, xo[dst]] @ W3[515] + b3
__global__ void lin3_kernel(const float* __restrict__ xm, const float* __restrict__ xo,
                            const float* __restrict__ ea,
                            const int* __restrict__ src, const int* __restrict__ dst,
                            const float* __restrict__ W3, const float* __restrict__ b3,
                            float* __restrict__ out, int E) {
  int wid = blockIdx.x * (blockDim.x >> 5) + (threadIdx.x >> 5);
  int lane = threadIdx.x & 31;
  if (wid >= E) return;
  int s = src[wid], d = dst[wid];
  const float* pm = xm + (size_t)s * HC + lane * 8;
  const float* po = xo + (size_t)d * HC + lane * 8;
  const float* wm = W3 + lane * 8;
  const float* wo = W3 + 259 + lane * 8;
  float sum = 0.f;
#pragma unroll
  for (int j = 0; j < 8; ++j) sum += pm[j] * wm[j] + po[j] * wo[j];
#pragma unroll
  for (int o = 1; o < 32; o <<= 1) sum += __shfl_xor(sum, o, 32);
  if (lane == 0) {
    sum += ea[(size_t)wid * 3 + 0] * W3[256] + ea[(size_t)wid * 3 + 1] * W3[257] +
           ea[(size_t)wid * 3 + 2] * W3[258] + b3[0];
    out[wid] = sum;
  }
}

// ================================================================ launch
extern "C" void kernel_launch(void* const* d_in, const int* in_sizes, int n_in,
                              void* d_out, int out_size, void* d_ws, size_t ws_size,
                              hipStream_t stream) {
  const int NM = in_sizes[0] / 64;
  const int NO = in_sizes[1] / 64;
  const int E  = in_sizes[2];

  const float* x_machine   = (const float*)d_in[0];
  const float* x_operation = (const float*)d_in[1];
  const int*   src_exec    = (const int*)d_in[2];
  const int*   dst_exec    = (const int*)d_in[3];
  const float* ea_exec     = (const float*)d_in[4];
  const float* ea_rev      = (const float*)d_in[5];
  auto P = [&](int i) { return (const float*)d_in[i]; };
  // params flatten (insertion order):
  // 6:lin1.machine.W 7:.b 8:lin1.operation.W 9:.b
  // per conv dir: Ws,bs,Wd,bd,We,att,b
  // 10..16 convs0.exec  17..23 convs0.rev  24..30 convs1.exec  31..37 convs1.rev
  // 38:lin3.W 39:lin3.b

  // ---- workspace bump allocator (256B aligned)
  char* wsb = (char*)d_ws;
  size_t off = 0;
  auto alloc = [&](size_t bytes) {
    void* p = wsb + off;
    off += (bytes + 255) & ~(size_t)255;
    return p;
  };
  float* xm8    = (float*)alloc((size_t)NM * 8 * 4);
  float* xo8    = (float*)alloc((size_t)NO * 8 * 4);
  float* xmA    = (float*)alloc((size_t)NM * HC * 4);  // after layer 0
  float* xoA    = (float*)alloc((size_t)NO * HC * 4);
  float* xmB    = (float*)alloc((size_t)NM * HC * 4);  // after layer 1
  float* xoB    = (float*)alloc((size_t)NO * HC * 4);
  float* bufS   = (float*)alloc((size_t)NO * HC * 4);  // xs projection
  float* bufD   = (float*)alloc((size_t)NO * HC * 4);  // xd projection
  float* logits = (float*)alloc((size_t)E * 4 * 4);
  float* mx     = (float*)alloc((size_t)NO * 4 * 4);
  float* den    = (float*)alloc((size_t)NO * 4 * 4);
  (void)ws_size; (void)n_in; (void)out_size;

  auto cdiv = [](long a, long b) { return (unsigned)((a + b - 1) / b); };
  const int eb = (int)cdiv(E, 8);  // wave-per-edge blocks (8 waves/block)

  auto gemm = [&](const float* X, const float* W, const float* b, float* o, int M) {
    long waves = (long)cdiv(M, 16) * 4;   // (M/16) row tiles * 4 column groups
    gemm_bf16_wmma<<<cdiv(waves * 32, 256), 256, 0, stream>>>(X, W, b, o, M);
  };

  auto run_edges = [&](const float* xs, const float* xd, const float* ea,
                       const float* We, const float* att, const float* bconv,
                       const int* esrc, const int* edst, float* outbuf, int n_dst) {
    int segs = n_dst * 4;
    init_seg_kernel<<<cdiv(segs, 256), 256, 0, stream>>>(mx, den, segs);
    size_t tot = (size_t)n_dst * HC;
    init_out_kernel<<<cdiv((long)tot, 256), 256, 0, stream>>>(outbuf, bconv, tot);
    edge_logits_kernel<<<eb, 256, 0, stream>>>(xs, xd, ea, We, att, esrc, edst, logits, mx, E);
    edge_exp_kernel<<<cdiv((long)E * 4, 256), 256, 0, stream>>>(logits, mx, den, edst, E * 4);
    edge_scatter_kernel<<<eb, 256, 0, stream>>>(logits, den, xs, esrc, edst, outbuf, E);
  };

  // ---- stage 1: lin1
  lin1_kernel<<<cdiv(NM, 256), 256, 0, stream>>>(x_machine,   P(6), P(7), xm8, NM);
  lin1_kernel<<<cdiv(NO, 256), 256, 0, stream>>>(x_operation, P(8), P(9), xo8, NO);

  // ---- layer 0 (in_dim 8) : exec (machine->operation), dst = operation
  proj8_kernel<<<cdiv((long)NM * HC, 256), 256, 0, stream>>>(xm8, P(10), P(11), bufS, NM);
  proj8_kernel<<<cdiv((long)NO * HC, 256), 256, 0, stream>>>(xo8, P(12), P(13), bufD, NO);
  run_edges(bufS, bufD, ea_exec, P(14), P(15), P(16), src_exec, dst_exec, xoA, NO);
  // rev: gatv2(xo, xm, ea_rev, src=dst_exec, dst=src_exec), dst = machine
  proj8_kernel<<<cdiv((long)NO * HC, 256), 256, 0, stream>>>(xo8, P(17), P(18), bufS, NO);
  proj8_kernel<<<cdiv((long)NM * HC, 256), 256, 0, stream>>>(xm8, P(19), P(20), bufD, NM);
  run_edges(bufS, bufD, ea_rev, P(21), P(22), P(23), dst_exec, src_exec, xmA, NM);

  // ---- layer 1 (in_dim 256, WMMA) : exec
  gemm(xmA, P(24), P(25), bufS, NM);
  gemm(xoA, P(26), P(27), bufD, NO);
  run_edges(bufS, bufD, ea_exec, P(28), P(29), P(30), src_exec, dst_exec, xoB, NO);
  // rev
  gemm(xoA, P(31), P(32), bufS, NO);
  gemm(xmA, P(33), P(34), bufD, NM);
  run_edges(bufS, bufD, ea_rev, P(35), P(36), P(37), dst_exec, src_exec, xmB, NM);

  // ---- final tanh + actor head
  tanh_kernel<<<cdiv((long)NM * HC, 256), 256, 0, stream>>>(xmB, (size_t)NM * HC);
  tanh_kernel<<<cdiv((long)NO * HC, 256), 256, 0, stream>>>(xoB, (size_t)NO * HC);
  lin3_kernel<<<eb, 256, 0, stream>>>(xmB, xoB, ea_exec, src_exec, dst_exec,
                                      P(38), P(39), (float*)d_out, E);
}
